// HGT_1829656068174
// MI455X (gfx1250) — compile-verified, matrix operand order checked by reading, not run on hardware
//
#include <hip/hip_runtime.h>
#include <hip/hip_bf16.h>
#include <math.h>

// ---------------------------------------------------------------------------
// Types for CDNA5 WMMA (wave32): v_wmma_f32_16x16x32_bf16
// ---------------------------------------------------------------------------
typedef __bf16 v16bf __attribute__((ext_vector_type(16)));
typedef __bf16 v8bf  __attribute__((ext_vector_type(8)));
typedef __bf16 v4bf  __attribute__((ext_vector_type(4)));
typedef float  v8f   __attribute__((ext_vector_type(8)));

#define HID 128
#define HEADS 4
#define DHEAD 32
#define LDSW 136   // padded LDS row stride in halfs (272B = 68 dwords):
                   // lane l of a fragment load starts at bank (4*l)%64 ->
                   // 16 lanes tile all 64 banks, conflict-free b128 loads.

// ---------------------------------------------------------------------------
// Monotonic float <-> uint encoding so segment-max can use integer atomicMax.
// ---------------------------------------------------------------------------
__device__ __forceinline__ unsigned fenc(float f) {
    unsigned u = __float_as_uint(f);
    return (u & 0x80000000u) ? ~u : (u | 0x80000000u);
}
__device__ __forceinline__ float fdec(unsigned k) {
    unsigned u = (k & 0x80000000u) ? (k & 0x7fffffffu) : ~k;
    return __uint_as_float(u);
}
// fenc(-inf) == 0x007FFFFF  (fill value for the running-max buffer)
#define NEG_INF_KEY 0x007FFFFFu

__device__ __forceinline__ float gelu_exact(float x) {
    return 0.5f * x * (1.0f + erff(x * 0.70710678118654752f));
}

// Assemble a 32B WMMA fragment from two 16B-aligned LDS halves
// (rows are 272B apart, so only 16B alignment is guaranteed).
__device__ __forceinline__ v16bf ldfrag(const __bf16* p) {
    union { v16bf v; v8bf h[2]; } u;
    u.h[0] = *(const v8bf*)p;
    u.h[1] = *(const v8bf*)(p + 8);
    return u.v;
}

// ---------------------------------------------------------------------------
// Generic fill (bit-pattern) kernel: used for zeroing and -inf init.
// ---------------------------------------------------------------------------
__global__ void fill_u32_kernel(unsigned* __restrict__ p, unsigned v, long long n) {
    long long i = (long long)blockIdx.x * blockDim.x + threadIdx.x;
    if (i < n) p[i] = v;
}

// ---------------------------------------------------------------------------
// Compose relation weights:  Wc = W @ blockdiag(rel),  bc = b @ blockdiag(rel)
//   rel is [H][D][D] (a_rel[l,r] or m_rel[l,r]);  head = j>>5, e = j&31
//   Wc[i,j] = sum_d W[i, head*32+d] * rel[head, d, e]
// ---------------------------------------------------------------------------
__global__ void compose_kernel(const float* __restrict__ W, const float* __restrict__ b,
                               const float* __restrict__ rel,
                               float* __restrict__ Wc, float* __restrict__ bc) {
    int idx = blockIdx.x * blockDim.x + threadIdx.x;
    if (idx < HID * HID) {
        int i = idx >> 7, j = idx & 127;
        int head = j >> 5, e = j & 31;
        const float* wrow = W + (size_t)i * HID + head * DHEAD;
        const float* rcol = rel + head * (DHEAD * DHEAD) + e;
        float s = 0.f;
#pragma unroll
        for (int d = 0; d < DHEAD; ++d) s += wrow[d] * rcol[d * DHEAD];
        Wc[idx] = s;
    } else if (idx < HID * HID + HID) {
        int j = idx - HID * HID;
        int head = j >> 5, e = j & 31;
        float s = 0.f;
#pragma unroll
        for (int d = 0; d < DHEAD; ++d)
            s += b[head * DHEAD + d] * rel[head * (DHEAD * DHEAD) + d * DHEAD + e];
        bc[j] = s;
    }
}

// ---------------------------------------------------------------------------
// Fused 128-wide GEMM on CDNA5 matrix cores:
//   out[r,:] = epilogue( act( (preGelu? gelu(A) : A)[r,:] @ W + bias ) )
// 256 threads = 8 wave32; wave w computes rows [tile*128 + 16w, +16) x all 128
// cols as 8 WMMA tiles.  A tile + W^T staged in LDS as bf16 (~68 KB, padded).
// Fragment layout used here (contiguous K halves, same order for A and B):
//   A (16x32 bf16): lane%16 = row M, lane/16 selects K half, 16 contiguous K.
//   B (32x16 bf16): lane%16 = col N, lane/16 selects K half -> load from W^T.
//   C/D (16x16 f32): VGPR r -> M = (lane/16)*8 + r, N = lane%16.
// B-fragments of each K-step are batch-loaded ahead of the 8-WMMA burst so
// the scheduler can issue 16 ds_load_b128 back-to-back with partial dscnt
// waits instead of a full drain per WMMA.
// Epilogue (optional): hn = sigmoid(skip)*a + (1-sigmoid(skip))*hprev (+hprev)
// ---------------------------------------------------------------------------
__global__ __launch_bounds__(256)
void gemm128_bf16_wmma(const float* __restrict__ A, const float* __restrict__ W,
                       const float* __restrict__ bias, float* __restrict__ out,
                       int nrows, int preGelu, int actRelu,
                       const float* __restrict__ hprev,
                       const float* __restrict__ skipPtr, int addRes) {
    __shared__ __bf16 Alds[128 * LDSW];   // row-major A tile (bf16, padded)
    __shared__ __bf16 Wlds[HID * LDSW];   // Wlds[n][k] = W[k][n] (W^T, padded)

    const int tid = threadIdx.x;
    const int rowBase = blockIdx.x * 128;

    // Stage W^T (bf16): global-coalesced f32 reads, scattered b16 LDS writes
    for (int i = tid; i < HID * HID; i += 256) {
        int k = i >> 7, n = i & 127;
        Wlds[n * LDSW + k] = (__bf16)W[i];
    }
    // Stage A tile (bf16): float4 global loads, v4bf LDS stores,
    // optional exact-GELU pre-activation.
    for (int i = tid; i < 128 * 32; i += 256) {
        int r = i >> 5;
        int c4 = (i & 31) << 2;
        int row = rowBase + r;
        float4 x = make_float4(0.f, 0.f, 0.f, 0.f);
        if (row < nrows) x = *(const float4*)(A + (size_t)row * HID + c4);
        if (preGelu) {
            x.x = gelu_exact(x.x); x.y = gelu_exact(x.y);
            x.z = gelu_exact(x.z); x.w = gelu_exact(x.w);
        }
        v4bf o;
        o[0] = (__bf16)x.x; o[1] = (__bf16)x.y;
        o[2] = (__bf16)x.z; o[3] = (__bf16)x.w;
        *(v4bf*)&Alds[r * LDSW + c4] = o;
    }
    __syncthreads();

    const int wave = tid >> 5;
    const int lane = tid & 31;
    const int lrow = lane & 15;            // M (A) / N (B) within tile
    const int lk   = (lane >> 4) << 4;     // K half: 0 or 16

    v8f acc[8];
#pragma unroll
    for (int j = 0; j < 8; ++j) acc[j] = (v8f)0.0f;

    // Hoist all 4 A fragments (rows fixed per wave)
    v16bf afr[4];
#pragma unroll
    for (int kk = 0; kk < 4; ++kk)
        afr[kk] = ldfrag(&Alds[(wave * 16 + lrow) * LDSW + kk * 32 + lk]);

#pragma unroll
    for (int kk = 0; kk < 4; ++kk) {
        v16bf bfr[8];
#pragma unroll
        for (int j = 0; j < 8; ++j)
            bfr[j] = ldfrag(&Wlds[(j * 16 + lrow) * LDSW + kk * 32 + lk]);
#pragma unroll
        for (int j = 0; j < 8; ++j)
            acc[j] = __builtin_amdgcn_wmma_f32_16x16x32_bf16(
                false, afr[kk], false, bfr[j], (short)0, acc[j], false, false);
    }

    // Epilogue
    const int rbase = rowBase + wave * 16 + ((lane >> 4) << 3);
    const int coll  = lane & 15;
    float beta = 0.f;
    if (hprev) beta = 1.0f / (1.0f + __expf(-skipPtr[0]));

#pragma unroll
    for (int j = 0; j < 8; ++j) {
        int n = j * 16 + coll;
        float bv = bias ? bias[n] : 0.0f;
#pragma unroll
        for (int r = 0; r < 8; ++r) {
            int row = rbase + r;
            if (row < nrows) {
                float val = acc[j][r] + bv;
                if (actRelu) val = fmaxf(val, 0.0f);
                if (hprev) {
                    float hp = hprev[(size_t)row * HID + n];
                    val = beta * val + (1.0f - beta) * hp;
                    if (addRes) val += hp;
                }
                out[(size_t)row * HID + n] = val;
            }
        }
    }
}

// ---------------------------------------------------------------------------
// Edge pass 1: logits[e,h] = scale * p_rel[h] * <q[dst,h,:], kr[src,h,:]>
//              + running segment-max into mkey[dst,h] (ordered-uint atomicMax)
// ---------------------------------------------------------------------------
__global__ void edge_logits_kernel(const int* __restrict__ edges,   // [2,E]
                                   const float* __restrict__ q,
                                   const float* __restrict__ kr,
                                   const float* __restrict__ prel,  // [H]
                                   float* __restrict__ logits,      // [E,H]
                                   unsigned* __restrict__ mkey,     // [N,H]
                                   int nE) {
    int t = blockIdx.x * blockDim.x + threadIdx.x;
    if (t >= nE * HEADS) return;
    int e = t >> 2, h = t & 3;
    int src = edges[e], dst = edges[nE + e];
    const float4* qp = (const float4*)(q  + (size_t)dst * HID + h * DHEAD);
    const float4* kp = (const float4*)(kr + (size_t)src * HID + h * DHEAD);
    float s = 0.f;
#pragma unroll
    for (int i = 0; i < 8; ++i) {
        float4 a = qp[i], b = kp[i];
        s += a.x * b.x + a.y * b.y + a.z * b.z + a.w * b.w;
    }
    s *= prel[h] * 0.17677669529663689f;   // 1/sqrt(32)
    logits[t] = s;
    atomicMax(mkey + (size_t)dst * HEADS + h, fenc(s));
}

// ---------------------------------------------------------------------------
// Edge pass 2: e = exp(logit - max) (in-place) + segment-sum into z[dst,h]
// ---------------------------------------------------------------------------
__global__ void edge_exp_kernel(const int* __restrict__ edges,
                                float* __restrict__ logits,
                                const unsigned* __restrict__ mkey,
                                float* __restrict__ z, int nE) {
    int t = blockIdx.x * blockDim.x + threadIdx.x;
    if (t >= nE * HEADS) return;
    int e = t >> 2, h = t & 3;
    int dst = edges[nE + e];
    float mm = fdec(mkey[(size_t)dst * HEADS + h]);
    if (mm < -3.0e38f) mm = 0.0f;          // isolated node: max stays 0
    float ex = __expf(logits[t] - mm);
    logits[t] = ex;
    atomicAdd(z + (size_t)dst * HEADS + h, ex);
}

// ---------------------------------------------------------------------------
// Edge pass 3: agg[dst,h,:] += (e / (z[dst,h]+eps)) * vr[src,h,:]
// ---------------------------------------------------------------------------
__global__ void edge_scatter_kernel(const int* __restrict__ edges,
                                    const float* __restrict__ logits,
                                    const float* __restrict__ z,
                                    const float* __restrict__ vr,
                                    float* __restrict__ agg, int nE) {
    int t = blockIdx.x * blockDim.x + threadIdx.x;
    if (t >= nE * HEADS) return;
    int e = t >> 2, h = t & 3;
    int src = edges[e], dst = edges[nE + e];
    float alpha = logits[t] / (z[(size_t)dst * HEADS + h] + 1e-16f);
    const float4* vp = (const float4*)(vr + (size_t)src * HID + h * DHEAD);
    float* ap = agg + (size_t)dst * HID + h * DHEAD;
#pragma unroll
    for (int i = 0; i < 8; ++i) {
        float4 v = vp[i];
        atomicAdd(ap + 4 * i + 0, alpha * v.x);
        atomicAdd(ap + 4 * i + 1, alpha * v.y);
        atomicAdd(ap + 4 * i + 2, alpha * v.z);
        atomicAdd(ap + 4 * i + 3, alpha * v.w);
    }
}

// ---------------------------------------------------------------------------
// Host orchestration
// ---------------------------------------------------------------------------
extern "C" void kernel_launch(void* const* d_in, const int* in_sizes, int n_in,
                              void* d_out, int out_size, void* d_ws, size_t ws_size,
                              hipStream_t stream) {
    const float* x_user = (const float*)d_in[0];
    const float* x_item = (const float*)d_in[1];
    const float* W_in   = (const float*)d_in[2];
    const float* b_in   = (const float*)d_in[3];
    const float* Wk     = (const float*)d_in[4];
    const float* bk     = (const float*)d_in[5];
    const float* Wq     = (const float*)d_in[6];
    const float* bq     = (const float*)d_in[7];
    const float* Wv     = (const float*)d_in[8];
    const float* bv     = (const float*)d_in[9];
    const float* a_rel  = (const float*)d_in[10];
    const float* m_rel  = (const float*)d_in[11];
    const float* p_rel  = (const float*)d_in[12];
    const float* Wo     = (const float*)d_in[13];
    const float* bo     = (const float*)d_in[14];
    const float* skip   = (const float*)d_in[15];
    const int* edge_ui  = (const int*)d_in[16];
    const int* edge_iu  = (const int*)d_in[17];

    const int Nn = in_sizes[0] / HID;      // nodes per type
    const int Ee = in_sizes[16] / 2;       // edges per relation
    const int L = 2, T = 2, R = 2;
    const size_t NH = (size_t)Nn * HID;

    // Workspace layout (f32)
    float* ws = (float*)d_ws;
    float* h[2]   = { ws,           ws + NH };
    float* q[2]   = { ws + 2 * NH,  ws + 3 * NH };
    float* kr     = ws + 4 * NH;
    float* vr     = ws + 5 * NH;
    float* agg[2] = { ws + 6 * NH,  ws + 7 * NH };
    float* logits = ws + 8 * NH;                                // [E,H]
    unsigned* mkey = (unsigned*)(logits + (size_t)Ee * HEADS);  // [N,H]
    float* z      = (float*)(mkey + (size_t)Nn * HEADS);        // [N,H]
    float* Wck    = z + (size_t)Nn * HEADS;
    float* bck    = Wck + HID * HID;
    float* Wcv    = bck + HID;
    float* bcv    = Wcv + HID * HID;

    const dim3 blk(256);
    const int gGemm = (Nn + 127) / 128;
    const int gEdge = (Ee * HEADS + 255) / 256;
    const int gNH   = (int)((NH + 255) / 256);
    const int gNHd  = (Nn * HEADS + 255) / 256;
    const int gComp = (HID * HID + HID + 255) / 256;

    auto gemm = [&](const float* A, const float* W, const float* b, float* o,
                    int preGelu, int actRelu, const float* hp, const float* sp,
                    int addRes) {
        gemm128_bf16_wmma<<<gGemm, blk, 0, stream>>>(A, W, b, o, Nn, preGelu,
                                                     actRelu, hp, sp, addRes);
    };

    // 1) Input projections: h[t] = relu(x @ W_in[t] + b_in[t])
    gemm(x_user, W_in,             b_in,       h[0], 0, 1, nullptr, nullptr, 0);
    gemm(x_item, W_in + HID * HID, b_in + HID, h[1], 0, 1, nullptr, nullptr, 0);

    const int* edges[2] = { edge_ui, edge_iu };
    const int stv[2] = { 0, 1 }, dtv[2] = { 1, 0 };

    for (int l = 0; l < L; ++l) {
        // 2a) Q projections per node type
        for (int t = 0; t < T; ++t) {
            size_t wo = (size_t)(l * T + t) * HID * HID;
            size_t bo_ = (size_t)(l * T + t) * HID;
            gemm(h[t], Wq + wo, bq + bo_, q[t], 0, 0, nullptr, nullptr, 0);
        }
        // 2b) zero aggregation buffers
        fill_u32_kernel<<<gNH, blk, 0, stream>>>((unsigned*)agg[0], 0u, (long long)NH);
        fill_u32_kernel<<<gNH, blk, 0, stream>>>((unsigned*)agg[1], 0u, (long long)NH);

        // 2c) per-relation attention
        for (int r = 0; r < R; ++r) {
            int st = stv[r], dt = dtv[r];
            size_t wko = (size_t)(l * T + st) * HID * HID;
            size_t bko = (size_t)(l * T + st) * HID;
            size_t ro  = (size_t)(l * R + r) * HEADS * DHEAD * DHEAD;

            // Composed weights: fold per-head a_rel/m_rel into Wk/Wv
            compose_kernel<<<gComp, blk, 0, stream>>>(Wk + wko, bk + bko,
                                                      a_rel + ro, Wck, bck);
            compose_kernel<<<gComp, blk, 0, stream>>>(Wv + wko, bv + bko,
                                                      m_rel + ro, Wcv, bcv);
            // k_r, v_r node tables via single WMMA GEMM each
            gemm(h[st], Wck, bck, kr, 0, 0, nullptr, nullptr, 0);
            gemm(h[st], Wcv, bcv, vr, 0, 0, nullptr, nullptr, 0);

            // segment softmax + scatter
            fill_u32_kernel<<<gNHd, blk, 0, stream>>>(mkey, NEG_INF_KEY,
                                                      (long long)Nn * HEADS);
            fill_u32_kernel<<<gNHd, blk, 0, stream>>>((unsigned*)z, 0u,
                                                      (long long)Nn * HEADS);
            edge_logits_kernel<<<gEdge, blk, 0, stream>>>(
                edges[r], q[dt], kr, p_rel + (size_t)(l * R + r) * HEADS,
                logits, mkey, Ee);
            edge_exp_kernel<<<gEdge, blk, 0, stream>>>(edges[r], logits, mkey, z, Ee);
            edge_scatter_kernel<<<gEdge, blk, 0, stream>>>(edges[r], logits, z,
                                                           vr, agg[dt], Ee);
        }

        // 2d) output projection + gated skip (+ residual for l>0)
        for (int t = 0; t < T; ++t) {
            size_t wo = (size_t)(l * T + t) * HID * HID;
            size_t bo_ = (size_t)(l * T + t) * HID;
            float* dst = (l == L - 1) ? ((float*)d_out + (size_t)t * NH) : h[t];
            gemm(agg[t], Wo + wo, bo + bo_, dst,
                 /*preGelu=*/1, /*actRelu=*/0,
                 /*hprev=*/h[t], /*skip=*/skip + (size_t)(l * T + t),
                 /*addRes=*/(l > 0) ? 1 : 0);
        }
    }
}
